// GraphConvClassifier_43499428774456
// MI455X (gfx1250) — matrix operand order repaired and checked by reading, not compiled, not run on hardware
//
#include <hip/hip_runtime.h>
#include <hip/hip_bf16.h>
#include <math.h>

typedef __bf16 v16bf __attribute__((ext_vector_type(16)));
typedef __bf16 v8bf  __attribute__((ext_vector_type(8)));
typedef __bf16 v4bf  __attribute__((ext_vector_type(4)));
typedef float  v8f   __attribute__((ext_vector_type(8)));

constexpr int Nn = 100000;   // nodes
constexpr int Ee = 600000;   // edges
constexpr int Dd = 128;      // embed dim
constexpr int Gg = 1000;     // graphs
constexpr int Ll = 3;        // layers

// ---------- order-preserving float<->uint for atomic max ----------
__device__ __forceinline__ unsigned encodeF(float f) {
  unsigned u = __float_as_uint(f);
  return (u & 0x80000000u) ? ~u : (u | 0x80000000u);
}
__device__ __forceinline__ float decodeF(unsigned u) {
  return (u & 0x80000000u) ? __uint_as_float(u & 0x7FFFFFFFu) : __uint_as_float(~u);
}

// ---------- degree / normalization precompute ----------
__global__ void init_deg_kernel(float* deg, int n) {
  int i = blockIdx.x * blockDim.x + threadIdx.x;
  if (i < n) deg[i] = 1.0f;  // self loop
}
__global__ void count_deg_kernel(const int* __restrict__ dst, float* deg, int e) {
  int i = blockIdx.x * blockDim.x + threadIdx.x;
  if (i < e) atomicAdd(&deg[dst[i]], 1.0f);
}
__global__ void dinv_kernel(float* deg, int n) {
  int i = blockIdx.x * blockDim.x + threadIdx.x;
  if (i < n) deg[i] = rsqrtf(deg[i]);   // in place: deg -> dinv
}
__global__ void enorm_kernel(const int* __restrict__ src, const int* __restrict__ dst,
                             const float* __restrict__ dinv, float* __restrict__ en, int e) {
  int i = blockIdx.x * blockDim.x + threadIdx.x;
  if (i < e) en[i] = dinv[src[i]] * dinv[dst[i]];
}

// Column-tile sweep: 8 tiles of 16 cols; FULL => unguarded stores.
template <bool HAS_BIAS, bool RELU_OUT, bool FULL>
__device__ __forceinline__ void gemm_cols(
    const v16bf (&afrag)[4], const __bf16 (*sBt)[128],
    const float* __restrict__ bias, float* __restrict__ out,
    int row0, int n_rows, int wave, int g, int m)
{
  const int rbase = row0 + wave * 16 + 8 * g;   // C/D layout: VGPR r -> M = r + 8g
  for (int j = 0; j < 8; ++j) {
    // B k order here (contiguous, as A): lanes 0-15 hold K=kk..kk+15, lanes 16-31 hold K=kk+16..kk+31
    const __bf16* colp = &sBt[j * 16 + m][16 * g];
    v16bf bfrag[4];
#pragma unroll
    for (int kt = 0; kt < 4; ++kt)
      bfrag[kt] = *reinterpret_cast<const v16bf*>(colp + kt * 32);

    v8f acc = {};
#pragma unroll
    for (int kt = 0; kt < 4; ++kt)
      acc = __builtin_amdgcn_wmma_f32_16x16x32_bf16(false, afrag[kt], false, bfrag[kt],
                                                    (short)0, acc, false, false);

    const int col = j * 16 + m;
    float bcol = 0.f;
    if constexpr (HAS_BIAS) bcol = bias[col];   // one load per tile
    float* outp = out + (size_t)rbase * Dd + col;
#pragma unroll
    for (int r = 0; r < 8; ++r) {
      if (FULL || (rbase + r < n_rows)) {
        float v = acc[r];
        if constexpr (HAS_BIAS) v += bcol;
        if constexpr (RELU_OUT) v = fmaxf(v, 0.f);
        outp[(size_t)r * Dd] = v;
      }
    }
  }
}

// ---------- 128-col GEMM: out[N,128] = (relu?)(in[N,128]) @ W[128,128] (+bias)(relu?) ----------
// One block = 128 rows; 8 waves; each wave owns a 16-row fragment, sweeps 8 column tiles.
// LDS layouts chosen so every WMMA fragment is a contiguous per-lane load:
//   sA  [row][k]  -> A 16x32 frag = two 16B runs per lane  (2x ds_load_b128)
//   sBt [col][k]  -> B 32x16 frag = one 32B run per lane   (2x ds_load_b128)
template <bool RELU_IN, bool HAS_BIAS, bool RELU_OUT>
__global__ __launch_bounds__(256) void gemm128_bf16_kernel(
    const float* __restrict__ in, const float* __restrict__ W,
    const float* __restrict__ bias, float* __restrict__ out, int n_rows)
{
  __shared__ __bf16 sA[128][128];   // [row][k]
  __shared__ __bf16 sBt[128][128];  // [col][k]  (W transposed)

  const int row0 = blockIdx.x * 128;
  const int tid  = threadIdx.x;

  // cooperative staging: 4096 float4 slots per matrix
  for (int i = tid; i < 128 * 32; i += 256) {
    const int r = i >> 5;          // 0..127
    const int c = (i & 31) << 2;   // 0,4,...,124
    float4 va = make_float4(0.f, 0.f, 0.f, 0.f);
    const int row = row0 + r;
    if (row < n_rows) va = *reinterpret_cast<const float4*>(in + (size_t)row * Dd + c);
    if constexpr (RELU_IN) {
      va.x = fmaxf(va.x, 0.f); va.y = fmaxf(va.y, 0.f);
      va.z = fmaxf(va.z, 0.f); va.w = fmaxf(va.w, 0.f);
    }
    v4bf pa = { (__bf16)va.x, (__bf16)va.y, (__bf16)va.z, (__bf16)va.w };
    *reinterpret_cast<v4bf*>(&sA[r][c]) = pa;          // one ds_store_b64

    float4 vb = *reinterpret_cast<const float4*>(W + (size_t)r * Dd + c);
    sBt[c + 0][r] = (__bf16)vb.x;                      // transposed scatter
    sBt[c + 1][r] = (__bf16)vb.y;
    sBt[c + 2][r] = (__bf16)vb.z;
    sBt[c + 3][r] = (__bf16)vb.w;
  }
  __syncthreads();

  const int wave = tid >> 5;       // 0..7 -> 16-row fragment
  const int lane = tid & 31;
  const int g    = lane >> 4;      // lane half
  const int m    = lane & 15;

  // A fragments for all 4 k-tiles (independent of column tile)
  const __bf16* arowp = &sA[wave * 16 + m][0];
  v16bf afrag[4];
#pragma unroll
  for (int kt = 0; kt < 4; ++kt) {
    const int kk = kt * 32;
    // A k order here (contiguous, as B): lane-half g, elements 0..15 -> K=kk+16g..kk+16g+15
    v8bf lo8 = *reinterpret_cast<const v8bf*>(arowp + kk + 16 * g);
    v8bf hi8 = *reinterpret_cast<const v8bf*>(arowp + kk + 16 * g + 8);
    afrag[kt] = __builtin_shufflevector(lo8, hi8, 0, 1, 2, 3, 4, 5, 6, 7,
                                        8, 9, 10, 11, 12, 13, 14, 15);
  }

  // Hoist the uniform partial-block test out of the column loop.
  if (row0 + 128 <= n_rows)
    gemm_cols<HAS_BIAS, RELU_OUT, true >(afrag, sBt, bias, out, row0, n_rows, wave, g, m);
  else
    gemm_cols<HAS_BIAS, RELU_OUT, false>(afrag, sBt, bias, out, row0, n_rows, wave, g, m);
}

// ---------- agg = hW * dinv^2 (self loop) + bias ----------
__global__ void self_bias_kernel(const float* __restrict__ hW, const float* __restrict__ dinv,
                                 const float* __restrict__ bias, float* __restrict__ agg,
                                 int total) {
  int i = blockIdx.x * blockDim.x + threadIdx.x;
  if (i < total) {
    const int row = i >> 7;        // / 128
    const int col = i & 127;
    const float dn = dinv[row];
    agg[i] = hW[i] * dn * dn + bias[col];
  }
}

// ---------- edge scatter: agg[dst] += hW[src] * enorm  (32 lanes x float4 per edge) ----------
__global__ void edge_scatter_kernel(const float* __restrict__ hW, const int* __restrict__ src,
                                    const int* __restrict__ dst, const float* __restrict__ en,
                                    float* __restrict__ agg, int e_count) {
  const int t = blockIdx.x * blockDim.x + threadIdx.x;
  const int e = t >> 5;
  if (e >= e_count) return;
  const int c = (t & 31) << 2;
  const float w = en[e];
  const int s = src[e], d = dst[e];
  float4 v = *reinterpret_cast<const float4*>(hW + (size_t)s * Dd + c);
  float* p = agg + (size_t)d * Dd + c;
  atomicAdd(p + 0, v.x * w);
  atomicAdd(p + 1, v.y * w);
  atomicAdd(p + 2, v.z * w);
  atomicAdd(p + 3, v.w * w);
}

// ---------- segment stats init ----------
__global__ void seg_init_kernel(unsigned* m_u, float* z, int g) {
  int i = blockIdx.x * blockDim.x + threadIdx.x;
  if (i < g) { m_u[i] = 0x007FFFFFu /* encode(-inf) */; z[i] = 0.f; }
}

// ---------- per-node score (dot with lin2) + mask + segment max ----------
__global__ __launch_bounds__(256) void score_kernel(
    const float* __restrict__ t, const float* __restrict__ w2, const float* __restrict__ b2,
    const int* __restrict__ mask, const int* __restrict__ batch,
    float* __restrict__ s, unsigned* __restrict__ m_u, int n)
{
  const int node = blockIdx.x * 8 + (threadIdx.x >> 5);  // one wave per node
  const int lane = threadIdx.x & 31;
  if (node >= n) return;
  float4 tv = *reinterpret_cast<const float4*>(t + (size_t)node * Dd + lane * 4);
  float4 wv = *reinterpret_cast<const float4*>(w2 + lane * 4);
  float p = tv.x * wv.x + tv.y * wv.y + tv.z * wv.z + tv.w * wv.w;
  for (int off = 16; off > 0; off >>= 1) p += __shfl_xor(p, off, 32);
  if (lane == 0) {
    float sc = p + b2[0];
    if (mask[node] == 0) sc = -1e9f;
    s[node] = sc;
    atomicMax(&m_u[batch[node]], encodeF(sc));
  }
}

// ---------- exp(s - max) + segment sum ----------
__global__ void exp_kernel(const float* __restrict__ s, const int* __restrict__ batch,
                           const unsigned* __restrict__ m_u, float* __restrict__ z,
                           float* __restrict__ out, int n) {
  int i = blockIdx.x * blockDim.x + threadIdx.x;
  if (i < n) {
    const int b = batch[i];
    const float ev = expf(s[i] - decodeF(m_u[b]));
    atomicAdd(&z[b], ev);
    out[i] = ev;
  }
}
__global__ void norm_kernel(const int* __restrict__ batch, const float* __restrict__ z,
                            float* __restrict__ out, int n) {
  int i = blockIdx.x * blockDim.x + threadIdx.x;
  if (i < n) out[i] = out[i] / z[batch[i]];
}

extern "C" void kernel_launch(void* const* d_in, const int* in_sizes, int n_in,
                              void* d_out, int out_size, void* d_ws, size_t ws_size,
                              hipStream_t stream) {
  (void)in_sizes; (void)n_in; (void)out_size; (void)ws_size;
  const float* x      = (const float*)d_in[0];
  const int*   ei     = (const int*)d_in[1];
  const int*   src    = ei;
  const int*   dst    = ei + Ee;
  const int*   batch  = (const int*)d_in[2];
  const int*   mask   = (const int*)d_in[3];
  const float* gcn_w  = (const float*)d_in[4];
  const float* gcn_b  = (const float*)d_in[5];
  const float* lin1_w = (const float*)d_in[6];
  const float* lin1_b = (const float*)d_in[7];
  const float* lin2_w = (const float*)d_in[8];
  const float* lin2_b = (const float*)d_in[9];
  float* out = (float*)d_out;

  // workspace layout (floats)
  float*    ws   = (float*)d_ws;
  float*    hW   = ws;                         // [N,D] matmul output
  float*    agg  = hW + (size_t)Nn * Dd;       // [N,D] aggregation / next-layer input
  float*    dinv = agg + (size_t)Nn * Dd;      // [N]  deg -> rsqrt(deg)
  float*    en   = dinv + Nn;                  // [E]  edge norm
  unsigned* m_u  = (unsigned*)(en + Ee);       // [G]  encoded segment max
  float*    z    = (float*)(m_u + Gg);         // [G]  segment sum
  float*    s    = agg;                        // [N]  scores reuse agg (free after lin1 GEMM)

  const dim3 b256(256);
  init_deg_kernel <<<(Nn + 255) / 256, b256, 0, stream>>>(dinv, Nn);
  count_deg_kernel<<<(Ee + 255) / 256, b256, 0, stream>>>(dst, dinv, Ee);
  dinv_kernel     <<<(Nn + 255) / 256, b256, 0, stream>>>(dinv, Nn);
  enorm_kernel    <<<(Ee + 255) / 256, b256, 0, stream>>>(src, dst, dinv, en, Ee);

  const int gemm_blocks = (Nn + 127) / 128;
  for (int l = 0; l < Ll; ++l) {
    const float* hin = (l == 0) ? x : agg;
    if (l == 0)
      gemm128_bf16_kernel<false, false, false><<<gemm_blocks, b256, 0, stream>>>(
          hin, gcn_w + (size_t)l * Dd * Dd, nullptr, hW, Nn);
    else
      gemm128_bf16_kernel<true, false, false><<<gemm_blocks, b256, 0, stream>>>(
          hin, gcn_w + (size_t)l * Dd * Dd, nullptr, hW, Nn);
    self_bias_kernel<<<(Nn * Dd + 255) / 256, b256, 0, stream>>>(
        hW, dinv, gcn_b + (size_t)l * Dd, agg, Nn * Dd);
    edge_scatter_kernel<<<(Ee * 32 + 255) / 256, b256, 0, stream>>>(
        hW, src, dst, en, agg, Ee);
  }
  // MLP head: t = relu(h @ lin1_w + lin1_b)
  gemm128_bf16_kernel<false, true, true><<<gemm_blocks, b256, 0, stream>>>(
      agg, lin1_w, lin1_b, hW, Nn);

  seg_init_kernel<<<(Gg + 255) / 256, b256, 0, stream>>>(m_u, z, Gg);
  score_kernel   <<<(Nn + 7) / 8,    b256, 0, stream>>>(hW, lin2_w, lin2_b, mask, batch, s, m_u, Nn);
  exp_kernel     <<<(Nn + 255) / 256, b256, 0, stream>>>(s, batch, m_u, z, out, Nn);
  norm_kernel    <<<(Nn + 255) / 256, b256, 0, stream>>>(batch, z, out, Nn);
}
